// SingelHeadAttention_71098888618731
// MI455X (gfx1250) — matrix operand order repaired and checked by reading, not compiled, not run on hardware
//
#include <hip/hip_runtime.h>

#define N_EMBED 1024
#define HEAD    64
#define SEQ     4096
#define BATCH   4

typedef __attribute__((ext_vector_type(16))) __bf16 v16bf;
typedef __attribute__((ext_vector_type(8)))  float  v8f;
typedef __attribute__((ext_vector_type(4)))  unsigned u32x4;
typedef __attribute__((ext_vector_type(4)))  float    f32x4;
typedef unsigned short u16;

union Bf16x16 {
    u16   u[16];
    v16bf v;
    u32x4 q[2];
};

__device__ __forceinline__ u16 f2bf(float f) {
    unsigned u = __builtin_bit_cast(unsigned, f);
    u += 0x7FFFu + ((u >> 16) & 1u);   // round-to-nearest-even
    return (u16)(u >> 16);
}

__device__ __forceinline__ v8f wmma_bf16(v16bf a, v16bf b, v8f c) {
    return __builtin_amdgcn_wmma_f32_16x16x32_bf16(
        /*neg_a=*/false, a, /*neg_b=*/false, b,
        /*c_mod=*/(short)0, c, /*reuse_a=*/false, /*reuse_b=*/false);
}

// A-matrix 16x32 bf16 fragment (ISA 7.12.2): lane holds row M=lane&15;
// halves 0..7 = K off8+0..7, halves 8..15 = K 16+off8+0..7, off8=(lane>=16)*8.
__device__ __forceinline__ v16bf frag_a_bf16(const u16* row_base, int col, int lane) {
    int off8 = (lane & 16) ? 8 : 0;
    const u16* p = row_base + col + off8;
    Bf16x16 t;
    t.q[0] = *reinterpret_cast<const u32x4*>(p);
    t.q[1] = *reinterpret_cast<const u32x4*>(p + 16);
    return t.v;
}

// B-matrix 32x16 bf16 fragment: lane holds column N=lane&15;
// halves 0..7 = K off8+0..7, halves 8..15 = K 16+off8+0..7, off8=(lane>=16)*8.
// row_base points at row N of B^T (row-major in memory).
__device__ __forceinline__ v16bf frag_b_bf16(const u16* row_base, int col, int lane) {
    int off16 = (lane & 16) ? 8 : 0;
    const u16* p = row_base + col + off16;
    Bf16x16 t;
    t.q[0] = *reinterpret_cast<const u32x4*>(p);
    t.q[1] = *reinterpret_cast<const u32x4*>(p + 16);
    return t.v;
}

// ---------------- Kernel 0a: W -> W^T bf16 ----------------
__global__ __launch_bounds__(256) void prep_kernel(
        const float* __restrict__ Wq, const float* __restrict__ Wk,
        const float* __restrict__ Wv, u16* __restrict__ wt) {
    int idx = blockIdx.x * 256 + threadIdx.x;
    if (idx >= 3 * HEAD * N_EMBED) return;
    int m = idx >> 16;
    int rem = idx & 65535;
    int n = rem >> 10;
    int k = rem & 1023;
    const float* W = (m == 0) ? Wq : ((m == 1) ? Wk : Wv);
    wt[idx] = f2bf(W[k * HEAD + n]);   // Wt[m][n][k] = W[k][n]
}

// ---------------- Kernel 0b: x fp32 -> bf16 (streaming) ----------------
__global__ __launch_bounds__(256) void xcvt_kernel(
        const float* __restrict__ x, u16* __restrict__ xb) {
    size_t i = ((size_t)blockIdx.x * 256 + threadIdx.x) * 8;   // grid sized exactly
    f32x4 a = *reinterpret_cast<const f32x4*>(x + i);
    f32x4 b = *reinterpret_cast<const f32x4*>(x + i + 4);
    union { u16 u[8]; u32x4 q; } t;
    t.u[0] = f2bf(a.x); t.u[1] = f2bf(a.y); t.u[2] = f2bf(a.z); t.u[3] = f2bf(a.w);
    t.u[4] = f2bf(b.x); t.u[5] = f2bf(b.y); t.u[6] = f2bf(b.z); t.u[7] = f2bf(b.w);
    *reinterpret_cast<u32x4*>(xb + i) = t.q;
}

// ---------------- Kernel 1: QKV projection ----------------
// 12 waves/block: wave w -> matrix (w%3) of row sub-tile (w/3).
// Ping-pong bundles (unroll-by-2) remove the register-rotation copies while
// keeping next-chunk loads in flight under the current WMMAs.
struct Bundle { v16bf ax; v16bf f[4]; };

__device__ __forceinline__ Bundle load_bundle(const u16* xrow, const u16* wbase,
                                              int k0, int lane) {
    Bundle b;
    b.ax = frag_a_bf16(xrow, k0, lane);
    #pragma unroll
    for (int t = 0; t < 4; ++t)
        b.f[t] = frag_b_bf16(wbase + (size_t)t * 16 * N_EMBED, k0, lane);
    return b;
}

__device__ __forceinline__ void bundle_mm(v8f acc[4], const Bundle& b) {
    #pragma unroll
    for (int t = 0; t < 4; ++t)
        acc[t] = wmma_bf16(b.ax, b.f[t], acc[t]);
}

__global__ __launch_bounds__(384) void qkv_kernel(
        const u16* __restrict__ xb, const u16* __restrict__ wt,
        const float* __restrict__ bq, const float* __restrict__ bk,
        const float* __restrict__ bv,
        u16* __restrict__ qb, u16* __restrict__ kb, u16* __restrict__ vt) {
    const int lane = threadIdx.x & 31;
    const int w    = threadIdx.x >> 5;          // 0..11
    const int mtx  = w % 3;                     // 0=q 1=k 2=v
    const int sub  = w / 3;                     // 0..3
    const int tiles_per_b = SEQ / 64;           // 64
    const int b    = blockIdx.x / tiles_per_b;
    const int s0   = (blockIdx.x % tiles_per_b) * 64 + sub * 16;
    const int ncol = lane & 15;
    const int moff = (lane & 16) ? 8 : 0;

    const u16* xrow  = xb + ((size_t)b * SEQ + s0 + ncol) * N_EMBED;
    const u16* wbase = wt + (size_t)mtx * HEAD * N_EMBED + (size_t)ncol * N_EMBED;

    v8f acc[4];
    const v8f z = {0.f,0.f,0.f,0.f,0.f,0.f,0.f,0.f};
    #pragma unroll
    for (int t = 0; t < 4; ++t) acc[t] = z;

    // ping-pong double buffer over 32 K-chunks of 32
    Bundle b0 = load_bundle(xrow, wbase, 0, lane);
    for (int i = 0; i < 15; ++i) {
        int k0 = i * 64;
        Bundle b1 = load_bundle(xrow, wbase, k0 + 32, lane);
        bundle_mm(acc, b0);
        b0 = load_bundle(xrow, wbase, k0 + 64, lane);
        bundle_mm(acc, b1);
    }
    Bundle b1 = load_bundle(xrow, wbase, N_EMBED - 32, lane);  // chunk 31
    bundle_mm(acc, b0);                                        // chunk 30
    bundle_mm(acc, b1);

    const float* bias = (mtx == 0) ? bq : ((mtx == 1) ? bk : bv);
    #pragma unroll
    for (int t = 0; t < 4; ++t) {
        float bb = bias[t * 16 + ncol];
        #pragma unroll
        for (int r = 0; r < 8; ++r) {
            int M = r + moff;
            int srow = s0 + M;
            int col  = t * 16 + ncol;
            u16 val = f2bf(acc[t][r] + bb);
            if (mtx == 0)      qb[((size_t)b * SEQ + srow) * HEAD + col] = val;
            else if (mtx == 1) kb[((size_t)b * SEQ + srow) * HEAD + col] = val;
            else               vt[((size_t)b * HEAD + col) * SEQ + srow] = val; // V^T
        }
    }
}

// ---------------- Kernel 2: causal flash attention ----------------
__global__ __launch_bounds__(256) void attn_kernel(
        const u16* __restrict__ qb, const u16* __restrict__ kb,
        const u16* __restrict__ vt, float* __restrict__ out) {
    const int lane = threadIdx.x & 31;
    const int wv   = threadIdx.x >> 5;
    const int tiles_per_b = SEQ / 128;
    const int b    = blockIdx.x / tiles_per_b;
    const int q0   = (blockIdx.x % tiles_per_b) * 128 + wv * 16;
    const int ncol = lane & 15;
    const int moff = (lane & 16) ? 8 : 0;

    const u16* qbase = qb + (size_t)b * SEQ * HEAD;
    const u16* kbase = kb + (size_t)b * SEQ * HEAD;
    const u16* vbase = vt + (size_t)b * HEAD * SEQ;

    const u16* qrow = qbase + (size_t)(q0 + ncol) * HEAD;
    v16bf aq0 = frag_a_bf16(qrow, 0,  lane);
    v16bf aq1 = frag_a_bf16(qrow, 32, lane);

    const v8f z = {0.f,0.f,0.f,0.f,0.f,0.f,0.f,0.f};
    v8f o[4];
    #pragma unroll
    for (int t = 0; t < 4; ++t) o[t] = z;
    float m[8], l[8];
    #pragma unroll
    for (int r = 0; r < 8; ++r) { m[r] = -1e30f; l[r] = 0.f; }

    __shared__ u16 pb[8][16 * 32];          // per-wave P staging tile
    u16* myp = &pb[wv][0];

    const u16* vrow0 = vbase + (size_t)ncol * SEQ;   // t adds 16*SEQ

    for (int k0 = 0; k0 < q0 + 16; k0 += 32) {
        const u16* kr0 = kbase + (size_t)(k0 + ncol) * HEAD;
        const u16* kr1 = kbase + (size_t)(k0 + 16 + ncol) * HEAD;

        // prefetch next key tile's K and V cachelines (global_prefetch_b8)
        if (k0 + 32 < q0 + 16) {
            __builtin_prefetch(kbase + (size_t)(k0 + 32 + ncol) * HEAD, 0, 3);
            __builtin_prefetch(kbase + (size_t)(k0 + 48 + ncol) * HEAD, 0, 3);
            __builtin_prefetch(vrow0 + k0 + 32, 0, 3);
        }

        // K fragments first, V fragments second: score WMMAs can proceed with
        // V loads (issued later) still outstanding; V waits overlap softmax.
        v16bf kb00 = frag_b_bf16(kr0, 0,  lane);
        v16bf kb01 = frag_b_bf16(kr0, 32, lane);
        v16bf kb10 = frag_b_bf16(kr1, 0,  lane);
        v16bf kb11 = frag_b_bf16(kr1, 32, lane);
        v16bf vbf[4];
        #pragma unroll
        for (int t = 0; t < 4; ++t)
            vbf[t] = frag_b_bf16(vrow0 + (size_t)t * 16 * SEQ, k0, lane);

        // ---- scores S = Q * K^T for 32 keys
        v8f s0 = z, s1 = z;
        s0 = wmma_bf16(aq0, kb00, s0);
        s0 = wmma_bf16(aq1, kb01, s0);
        s1 = wmma_bf16(aq0, kb10, s1);
        s1 = wmma_bf16(aq1, kb11, s1);

        // ---- causal mask + scale + online softmax
        float mnew[8];
        #pragma unroll
        for (int r = 0; r < 8; ++r) {
            int qr = q0 + r + moff;
            float v0 = (k0 + ncol      <= qr) ? s0[r] * 0.125f : -1e30f;
            float v1 = (k0 + 16 + ncol <= qr) ? s1[r] * 0.125f : -1e30f;
            s0[r] = v0; s1[r] = v1;
            float rm = fmaxf(v0, v1);
            rm = fmaxf(rm, __shfl_xor(rm, 1, 32));
            rm = fmaxf(rm, __shfl_xor(rm, 2, 32));
            rm = fmaxf(rm, __shfl_xor(rm, 4, 32));
            rm = fmaxf(rm, __shfl_xor(rm, 8, 32));
            mnew[r] = fmaxf(m[r], rm);
        }

        #pragma unroll
        for (int r = 0; r < 8; ++r) {
            float a  = __expf(m[r] - mnew[r]);
            float p0 = __expf(s0[r] - mnew[r]);
            float p1 = __expf(s1[r] - mnew[r]);
            float rs = p0 + p1;
            rs += __shfl_xor(rs, 1, 32);
            rs += __shfl_xor(rs, 2, 32);
            rs += __shfl_xor(rs, 4, 32);
            rs += __shfl_xor(rs, 8, 32);
            l[r] = l[r] * a + rs;
            m[r] = mnew[r];
            #pragma unroll
            for (int t = 0; t < 4; ++t) o[t][r] *= a;
            int M = r + moff;
            myp[M * 32 + ncol]      = f2bf(p0);
            myp[M * 32 + 16 + ncol] = f2bf(p1);
        }
        asm volatile("s_wait_dscnt 0" ::: "memory");

        // ---- P (16x32) as A fragment from LDS, then O += P * V
        v16bf ap = frag_a_bf16(myp + (size_t)ncol * 32, 0, lane);
        #pragma unroll
        for (int t = 0; t < 4; ++t)
            o[t] = wmma_bf16(ap, vbf[t], o[t]);
    }

    // ---- normalize and store fp32
    #pragma unroll
    for (int t = 0; t < 4; ++t) {
        #pragma unroll
        for (int r = 0; r < 8; ++r) {
            int M = r + moff;
            out[((size_t)b * SEQ + q0 + M) * HEAD + t * 16 + ncol] = o[t][r] / l[r];
        }
    }
}

extern "C" void kernel_launch(void* const* d_in, const int* in_sizes, int n_in,
                              void* d_out, int out_size, void* d_ws, size_t ws_size,
                              hipStream_t stream) {
    (void)in_sizes; (void)n_in; (void)out_size; (void)ws_size;
    const float* x  = (const float*)d_in[0];
    const float* Wq = (const float*)d_in[1];
    const float* bq = (const float*)d_in[2];
    const float* Wk = (const float*)d_in[3];
    const float* bk = (const float*)d_in[4];
    const float* Wv = (const float*)d_in[5];
    const float* bv = (const float*)d_in[6];

    // ws layout (bf16): Wt[3][64][1024] | xb[B][S][1024] | q[B][S][64] | k[B][S][64] | Vt[B][64][S]
    u16* wt   = (u16*)d_ws;
    u16* xbuf = wt + (size_t)3 * HEAD * N_EMBED;
    u16* qbuf = xbuf + (size_t)BATCH * SEQ * N_EMBED;
    u16* kbuf = qbuf + (size_t)BATCH * SEQ * HEAD;
    u16* vtb  = kbuf + (size_t)BATCH * SEQ * HEAD;

    prep_kernel<<<(3 * HEAD * N_EMBED + 255) / 256, 256, 0, stream>>>(Wq, Wk, Wv, wt);
    xcvt_kernel<<<(int)((size_t)BATCH * SEQ * N_EMBED / 8 / 256), 256, 0, stream>>>(x, xbuf);
    qkv_kernel<<<BATCH * (SEQ / 64), 384, 0, stream>>>(xbuf, wt, bq, bk, bv, qbuf, kbuf, vtb);
    attn_kernel<<<BATCH * (SEQ / 128), 256, 0, stream>>>(qbuf, kbuf, vtb, (float*)d_out);
}
